// GaussianSelfAttention_5514738008938
// MI455X (gfx1250) — matrix operand order repaired and checked by reading, not compiled, not run on hardware
//
#include <hip/hip_runtime.h>

// Problem constants (from the reference)
#define BATCH   64
#define SEQ     197
#define DMODEL  768
#define PP      196
#define M_ROWS  (BATCH * SEQ)          // 12608
#define M_PAD   (99 * 128)             // 12672, padded so GEMM A-side needs no bounds checks
#define WELEMS  (DMODEL * DMODEL)      // 589824
#define PGRIDF  14.0f

typedef __attribute__((ext_vector_type(16))) __bf16       v16bf;
typedef __attribute__((ext_vector_type(8)))  float        v8f;
typedef __attribute__((ext_vector_type(4)))  unsigned int u32x4;

// round-to-nearest-even f32 -> bf16, two at a time, packed into one u32
__device__ __forceinline__ unsigned int pack2_bf16(float a, float b) {
    unsigned int ua = __float_as_uint(a); ua += 0x7FFFu + ((ua >> 16) & 1u);
    unsigned int ub = __float_as_uint(b); ub += 0x7FFFu + ((ub >> 16) & 1u);
    return (ua >> 16) | (ub & 0xFFFF0000u);
}

// Load a 16-element bf16 fragment as two contiguous 16B chunks from LDS
__device__ __forceinline__ v16bf frag_from_lds(const unsigned short* p0,
                                               const unsigned short* p1) {
    union { u32x4 u[2]; v16bf v; } t;
    t.u[0] = *(const u32x4*)p0;
    t.u[1] = *(const u32x4*)p1;
    return t.v;
}

// CDNA5 async memory->LDS copy (16B per lane), tracked by ASYNCcnt.
// Each lane supplies its own LDS byte offset (vdst) and 64-bit global address.
__device__ __forceinline__ void async_load_b128(unsigned lds_off, const void* gaddr) {
    asm volatile("global_load_async_to_lds_b128 %0, %1, off"
                 :: "v"(lds_off), "v"(gaddr)
                 : "memory");
}

// ---------------------------------------------------------------------------
// Kernel 0: one-shot f32 -> bf16 conversion of x (zero-padded to M_PAD rows)
// and Wq/Wk/Wv into the workspace.  8 elements (one 16B store) per thread.
// ---------------------------------------------------------------------------
#define XPAD_ELEMS (M_PAD * DMODEL)                       // 9,732,096
#define CVT_TOTAL  (XPAD_ELEMS + 3 * WELEMS)              // 11,501,568 (div by 8)

__global__ __launch_bounds__(256)
void convert_bf16_kernel(const float* __restrict__ x,
                         const float* __restrict__ Wq,
                         const float* __restrict__ Wk,
                         const float* __restrict__ Wv,
                         unsigned short* __restrict__ dst)
{
    const long long e = ((long long)blockIdx.x * 256 + threadIdx.x) * 8;
    if (e >= CVT_TOTAL) return;

    float f[8];
    if (e < XPAD_ELEMS) {
        const int row = (int)(e / DMODEL);
        if (row < M_ROWS) {
            const float4 a = *(const float4*)(x + e);
            const float4 b = *(const float4*)(x + e + 4);
            f[0]=a.x; f[1]=a.y; f[2]=a.z; f[3]=a.w;
            f[4]=b.x; f[5]=b.y; f[6]=b.z; f[7]=b.w;
        } else {
            #pragma unroll
            for (int i = 0; i < 8; ++i) f[i] = 0.f;
        }
    } else {
        long long e2 = e - XPAD_ELEMS;
        const float* W = (e2 < WELEMS) ? Wq : (e2 < 2 * WELEMS ? Wk : Wv);
        long long off = e2 % WELEMS;
        const float4 a = *(const float4*)(W + off);
        const float4 b = *(const float4*)(W + off + 4);
        f[0]=a.x; f[1]=a.y; f[2]=a.z; f[3]=a.w;
        f[4]=b.x; f[5]=b.y; f[6]=b.z; f[7]=b.w;
    }
    u32x4 o;
    o.x = pack2_bf16(f[0], f[1]);
    o.y = pack2_bf16(f[2], f[3]);
    o.z = pack2_bf16(f[4], f[5]);
    o.w = pack2_bf16(f[6], f[7]);
    *(u32x4*)(dst + e) = o;
}

// ---------------------------------------------------------------------------
// Kernel 1: QKV projection GEMM, bf16 WMMA, async double-buffered staging.
// out[m, n] = sum_k x[m,k] * W[n,k] + bias[n];  blockIdx.z selects q/k/v.
// Tile 128x128, K staged 32 at a time; 8 waves, 64x32 sub-tile per wave.
// ---------------------------------------------------------------------------
__global__ __launch_bounds__(256)
void qkv_gemm_kernel(const unsigned short* __restrict__ xbf,
                     const unsigned short* __restrict__ Wqb,
                     const unsigned short* __restrict__ Wkb,
                     const unsigned short* __restrict__ Wvb,
                     const float* __restrict__ bq, const float* __restrict__ bk,
                     const float* __restrict__ bv,
                     float* __restrict__ qo, float* __restrict__ ko,
                     float* __restrict__ vo)
{
    // two stages; per stage: A tile 4096 bf16 then B tile 4096 bf16
    __shared__ alignas(16) unsigned short lds[2 * 8192];

    const unsigned short* Wb;  const float* bias;  float* out;
    if      (blockIdx.z == 0) { Wb = Wqb; bias = bq; out = qo; }
    else if (blockIdx.z == 1) { Wb = Wkb; bias = bk; out = ko; }
    else                      { Wb = Wvb; bias = bv; out = vo; }

    const int tid  = threadIdx.x;
    const int lane = tid & 31;
    const int wid  = tid >> 5;
    const int r    = lane & 15;
    const int hi   = lane >> 4;

    const int m0 = blockIdx.y * 128;
    const int n0 = blockIdx.x * 128;
    const int wm = (wid & 1) * 64;
    const int wn = (wid >> 1) * 32;

    const unsigned lds0 = (unsigned)(size_t)(&lds[0]);   // LDS byte offset of buffer

    // This thread's two 16B chunks per tile: chunk c covers row c>>2, k-sub (c&3)*8,
    // and lands at LDS byte offset c*16 within the tile.
    const int c0 = tid;        // 0..255
    const int c1 = tid + 256;  // 256..511

    // issue the 4 async-b128 instructions (A x2, B x2) for K-chunk kk into stage st
    auto issue_stage = [&](int kk, int st) {
        const unsigned base = lds0 + (unsigned)(st * 16384);
        #pragma unroll
        for (int h = 0; h < 2; ++h) {
            const int c   = h ? c1 : c0;
            const int row = c >> 2;
            const int ko  = (c & 3) * 8;
            async_load_b128(base + (unsigned)(c * 16),
                            xbf + (size_t)(m0 + row) * DMODEL + kk + ko);
            async_load_b128(base + 8192u + (unsigned)(c * 16),
                            Wb + (size_t)(n0 + row) * DMODEL + kk + ko);
        }
    };

    v8f acc[4][2] = {};

    issue_stage(0, 0);

    for (int ks = 0; ks < 24; ++ks) {
        if (ks + 1 < 24) {
            issue_stage((ks + 1) * 32, (ks + 1) & 1);
            // 4 newest (next stage) may remain outstanding; stage ks has landed
            asm volatile("s_wait_asynccnt 0x4" ::: "memory");
        } else {
            asm volatile("s_wait_asynccnt 0x0" ::: "memory");
        }
        __syncthreads();                      // all waves' stage-ks data visible

        const unsigned short* As = lds + (ks & 1) * 8192;
        const unsigned short* Bs = As + 4096;

        // B fragments: lane r = N, elems 0..15 -> K = hi*16 .. hi*16+15
        v16bf bfr[2];
        #pragma unroll
        for (int fn = 0; fn < 2; ++fn) {
            const int n = wn + fn * 16 + r;
            bfr[fn] = frag_from_lds(&Bs[n * 32 + hi * 16],
                                    &Bs[n * 32 + hi * 16 + 8]);
        }
        // A fragments: lane r = M, elems 0..15 -> K = hi*16 .. hi*16+15
        #pragma unroll
        for (int fm = 0; fm < 4; ++fm) {
            const int m = wm + fm * 16 + r;
            v16bf afr = frag_from_lds(&As[m * 32 + hi * 16],
                                      &As[m * 32 + hi * 16 + 8]);
            #pragma unroll
            for (int fn = 0; fn < 2; ++fn) {
                acc[fm][fn] = __builtin_amdgcn_wmma_f32_16x16x32_bf16(
                    false, afr, false, bfr[fn],
                    (short)0, acc[fm][fn], false, false);
            }
        }
        __syncthreads();   // stage-ks buffer free for reuse (overwritten at ks+2)
    }

    // ---- store C + bias.  D layout: VGPR i -> row = base + hi*8 + i, col = base + r
    #pragma unroll
    for (int fn = 0; fn < 2; ++fn) {
        const int col = n0 + wn + fn * 16 + r;
        const float bv_ = bias[col];
        #pragma unroll
        for (int fm = 0; fm < 4; ++fm) {
            #pragma unroll
            for (int i = 0; i < 8; ++i) {
                const int row = m0 + wm + fm * 16 + hi * 8 + i;
                if (row < M_ROWS)
                    out[(size_t)row * DMODEL + col] = acc[fm][fn][i] + bv_;
            }
        }
    }
}

// ---------------------------------------------------------------------------
// Kernel 2: Gaussian gather + 4-way attention.  One wave32 per (b, s).
// ---------------------------------------------------------------------------
__global__ __launch_bounds__(256)
void gauss_attn_kernel(const float* __restrict__ q,
                       const float* __restrict__ k,
                       const float* __restrict__ v,
                       const int*   __restrict__ img_ids,
                       const float* __restrict__ noise,
                       const float* __restrict__ avgs,
                       const float* __restrict__ stds,
                       float* __restrict__ out)
{
    const int w    = blockIdx.x * 8 + (threadIdx.x >> 5);
    const int lane = threadIdx.x & 31;
    if (w >= M_ROWS) return;
    const int b = w / SEQ;
    const int s = w % SEQ;

    float* op = out + (size_t)w * DMODEL;

    if (s == 0) {   // class token: 4 equal scores over all-ones rows -> exactly 1.0
        #pragma unroll
        for (int j = 0; j < 24; ++j) op[lane + j * 32] = 1.0f;
        return;
    }

    const int p   = s - 1;
    const int img = img_ids[b];
    const float ax = avgs[((size_t)img * 2 + 0) * PP + p];
    const float ay = avgs[((size_t)img * 2 + 1) * PP + p];
    const float sx = stds[((size_t)img * 2 + 0) * PP + p];
    const float sy = stds[((size_t)img * 2 + 1) * PP + p];
    const float kx = (noise[b * 2 + 0] - ax) / sx;
    const float ky = (noise[b * 2 + 1] - ay) / sy;
    const float xc = ceilf(kx),  xf = floorf(kx);
    const float yc = ceilf(ky),  yf = floorf(ky);

    int idx[4];
    idx[0] = (int)(PGRIDF * yc + xc);
    idx[1] = (int)(PGRIDF * yc + xf);
    idx[2] = (int)(PGRIDF * yf + xc);
    idx[3] = (int)(PGRIDF * yf + xf);
    #pragma unroll
    for (int a = 0; a < 4; ++a) idx[a] = min(max(idx[a], 0), SEQ - 1);

    float qr[24];
    const float* qp = q + (size_t)w * DMODEL;
    #pragma unroll
    for (int j = 0; j < 24; ++j) qr[j] = qp[lane + j * 32];

    float sc[4];
    #pragma unroll
    for (int a = 0; a < 4; ++a) {
        const float* kp = k + ((size_t)b * SEQ + idx[a]) * DMODEL;
        float accd = 0.f;
        #pragma unroll
        for (int j = 0; j < 24; ++j) accd += qr[j] * kp[lane + j * 32];
        #pragma unroll
        for (int off = 16; off; off >>= 1) accd += __shfl_xor(accd, off, 32);
        sc[a] = accd;
    }

    const float mx = fmaxf(fmaxf(sc[0], sc[1]), fmaxf(sc[2], sc[3]));
    float e[4], den = 0.f;
    #pragma unroll
    for (int a = 0; a < 4; ++a) { e[a] = __expf(sc[a] - mx); den += e[a]; }
    const float inv = 1.f / den;
    float wt[4];
    #pragma unroll
    for (int a = 0; a < 4; ++a) wt[a] = e[a] * inv;

    const float* vp0 = v + ((size_t)b * SEQ + idx[0]) * DMODEL;
    const float* vp1 = v + ((size_t)b * SEQ + idx[1]) * DMODEL;
    const float* vp2 = v + ((size_t)b * SEQ + idx[2]) * DMODEL;
    const float* vp3 = v + ((size_t)b * SEQ + idx[3]) * DMODEL;
    #pragma unroll
    for (int j = 0; j < 24; ++j) {
        const int d = lane + j * 32;
        op[d] = wt[0] * vp0[d] + wt[1] * vp1[d] + wt[2] * vp2[d] + wt[3] * vp3[d];
    }
}

// ---------------------------------------------------------------------------
extern "C" void kernel_launch(void* const* d_in, const int* in_sizes, int n_in,
                              void* d_out, int out_size, void* d_ws, size_t ws_size,
                              hipStream_t stream)
{
    (void)in_sizes; (void)n_in; (void)out_size; (void)ws_size;

    const float* x       = (const float*)d_in[0];
    const int*   img_ids = (const int*)  d_in[1];
    // d_in[2] = mask (unused)
    const float* noise   = (const float*)d_in[3];
    const float* Wq      = (const float*)d_in[4];
    const float* bq      = (const float*)d_in[5];
    const float* Wk      = (const float*)d_in[6];
    const float* bk      = (const float*)d_in[7];
    const float* Wv      = (const float*)d_in[8];
    const float* bv      = (const float*)d_in[9];
    const float* avgs    = (const float*)d_in[10];
    const float* stds    = (const float*)d_in[11];
    float*       out     = (float*)d_out;

    // workspace layout: q | k | v (f32) | bf16 region (x padded, Wq, Wk, Wv)
    const size_t per = (size_t)M_ROWS * DMODEL;
    float* q = (float*)d_ws;
    float* k = q + per;
    float* v = k + per;
    unsigned short* bf  = (unsigned short*)(v + per);   // 16B-aligned (offset 116,195,328)
    unsigned short* xbf = bf;
    unsigned short* Wqb = bf + (size_t)XPAD_ELEMS;
    unsigned short* Wkb = Wqb + WELEMS;
    unsigned short* Wvb = Wkb + WELEMS;

    // 1) convert to bf16 (and zero-pad x rows up to M_PAD)
    const int cvt_threads = CVT_TOTAL / 8;               // 1,437,696
    convert_bf16_kernel<<<dim3((cvt_threads + 255) / 256), 256, 0, stream>>>(
        x, Wq, Wk, Wv, bf);

    // 2) QKV GEMMs: 6 N-tiles, 99 M-tiles, z = q/k/v
    dim3 gemm_grid(6, 99, 3);
    qkv_gemm_kernel<<<gemm_grid, 256, 0, stream>>>(xbf, Wqb, Wkb, Wvb,
                                                   bq, bk, bv, q, k, v);

    // 3) gather-attention: 12608 (b,s) pairs / 8 waves per block
    gauss_attn_kernel<<<dim3(1576), 256, 0, stream>>>(q, k, v, img_ids, noise,
                                                      avgs, stds, out);
}
